// get_model_833223655901
// MI455X (gfx1250) — matrix operand order repaired and checked by reading, not compiled, not run on hardware
//
#include <hip/hip_runtime.h>
#include <hip/hip_bf16.h>

// ===========================================================================
// PointNet++ part-seg forward for MI455X (gfx1250, wave32, WMMA + TDM).
// GEMMs run on v_wmma_f32_16x16x32_f16 (f16 in, f32 acc). Weight tiles are
// DMA-staged into LDS once per workgroup via tensor_load_to_lds (TDM), gated
// by s_wait_tensorcnt. BN is global per-channel over the whole batch
// (matching the JAX reference): GEMM -> stats reduction -> fused BN+ReLU
// that re-packs f16 for the next layer.
// ===========================================================================

typedef __attribute__((ext_vector_type(16))) _Float16 v16h;
typedef __attribute__((ext_vector_type(8)))  float    v8f;
typedef __attribute__((ext_vector_type(4)))  unsigned v4u;
typedef __attribute__((ext_vector_type(8)))  int      v8i;
typedef __attribute__((ext_vector_type(4)))  int      v4i;

#define BN_EPSF 1e-5f
static const int kB = 32;
static const int kN = 2048;
static const int kCls = 50;
static const int kCkMax = 1280;   // largest GEMM K-dim (FP3 layer 1)

// ---------------------------------------------------------------- transpose
__global__ void k_transpose(const float* __restrict__ in, float* __restrict__ out,
                            int b_, int n_) {
  long i = (long)blockIdx.x * blockDim.x + threadIdx.x;
  long tot = (long)b_ * n_ * 3;
  if (i >= tot) return;
  int c = (int)(i % 3);
  long r = i / 3;
  int p = (int)(r % n_);
  int b = (int)(r / n_);
  out[i] = in[(size_t)b * 3 * n_ + (size_t)c * n_ + p];
}

// ---------------------------------------------------------------- FPS
// One block per batch; dist[] lives in LDS (8KB of 320KB WGP pool).
__global__ __launch_bounds__(256) void k_fps(const float* __restrict__ xyz,
                                             int n, int npoint, int* __restrict__ out) {
  __shared__ float dist[2048];
  __shared__ float rv[256];
  __shared__ int   ri[256];
  int b = blockIdx.x;
  int t = threadIdx.x;
  const float* p = xyz + (size_t)b * n * 3;
  for (int i = t; i < n; i += 256) dist[i] = 1e10f;
  __syncthreads();
  int far = 0;
  for (int it = 0; it < npoint; ++it) {
    if (t == 0) out[(size_t)b * npoint + it] = far;
    float cx = p[far * 3 + 0], cy = p[far * 3 + 1], cz = p[far * 3 + 2];
    float best = -1.0f; int bi = 0;
    for (int i = t; i < n; i += 256) {
      float dx = p[i * 3 + 0] - cx;
      float dy = p[i * 3 + 1] - cy;
      float dz = p[i * 3 + 2] - cz;
      float d = dx * dx + dy * dy + dz * dz;
      float dm = fminf(dist[i], d);
      dist[i] = dm;
      if (dm > best) { best = dm; bi = i; }
    }
    rv[t] = best; ri[t] = bi;
    __syncthreads();
    for (int s = 128; s > 0; s >>= 1) {
      if (t < s && rv[t + s] > rv[t]) { rv[t] = rv[t + s]; ri[t] = ri[t + s]; }
      __syncthreads();
    }
    far = ri[0];
    __syncthreads();
  }
}

// ---------------------------------------------------------------- gather xyz
__global__ void k_gather_xyz(const float* __restrict__ xyz, const int* __restrict__ fidx,
                             float* __restrict__ out, int n, int S, int b_) {
  long i = (long)blockIdx.x * blockDim.x + threadIdx.x;
  long tot = (long)b_ * S * 3;
  if (i >= tot) return;
  int c = (int)(i % 3);
  long r = i / 3;
  int s = (int)(r % S);
  int b = (int)(r / S);
  int id = fidx[(size_t)b * S + s];
  out[i] = xyz[((size_t)b * n + id) * 3 + c];
}

// ---------------------------------------------------------------- ball query
// One wave32 per center; ballot + popc-prefix collects the first K in-radius
// point indices in ascending index order (== reference's sort+truncate+pad).
__global__ __launch_bounds__(256) void k_query_ball(const float* __restrict__ xyz,
                                                    const float* __restrict__ new_xyz,
                                                    int n, int S, float r2, int K,
                                                    int* __restrict__ idx) {
  __shared__ int buf[8][64];
  int lane = threadIdx.x & 31;
  int w    = threadIdx.x >> 5;
  long gw  = ((long)blockIdx.x * blockDim.x + threadIdx.x) >> 5;
  int b = (int)(gw / S);
  const float* q = new_xyz + (size_t)gw * 3;
  float cx = q[0], cy = q[1], cz = q[2];
  const float* p = xyz + (size_t)b * n * 3;
  int cnt = 0;
  for (int base = 0; base < n && cnt < K; base += 32) {
    int i = base + lane;
    float dx = p[i * 3 + 0] - cx;
    float dy = p[i * 3 + 1] - cy;
    float dz = p[i * 3 + 2] - cz;
    float d = dx * dx + dy * dy + dz * dz;
    bool pred = (i < n) && (d <= r2);
    unsigned mask = (unsigned)__ballot(pred);
    int pos = cnt + __popc(mask & ((1u << lane) - 1u));
    if (pred && pos < K) buf[w][pos] = i;
    cnt += __popc(mask);
  }
  __builtin_amdgcn_wave_barrier();
  if (cnt > K) cnt = K;
  int first = buf[w][0];                 // center itself (d==0) guarantees cnt>=1
  int* o = idx + (size_t)gw * K;
  for (int j = lane; j < K; j += 32) o[j] = (j < cnt) ? buf[w][j] : first;
}

// ---------------------------------------------------------------- grouping
// Build f16 A-matrix rows: [xyz - center (3) | point feats (cpts)] padded to Ck.
__global__ void k_group(const float* __restrict__ xyz, const float* __restrict__ pts,
                        int cpts, const int* __restrict__ qidx,
                        const float* __restrict__ new_xyz,
                        int n, int S, int K, int Ck, _Float16* __restrict__ A, int b_) {
  long tid = (long)blockIdx.x * blockDim.x + threadIdx.x;
  long tot = (long)b_ * S * K;
  if (tid >= tot) return;
  long bs = tid / K;
  int b = (int)(bs / S);
  int id = qidx[tid];
  const float* c = new_xyz + (size_t)bs * 3;
  const float* P = xyz + ((size_t)b * n + id) * 3;
  const float* F = pts + ((size_t)b * n + id) * cpts;
  _Float16* row = A + (size_t)tid * Ck;
  row[0] = (_Float16)(P[0] - c[0]);
  row[1] = (_Float16)(P[1] - c[1]);
  row[2] = (_Float16)(P[2] - c[2]);
  for (int ch = 0; ch < cpts; ++ch) row[3 + ch] = (_Float16)F[ch];
  for (int ch = 3 + cpts; ch < Ck; ++ch) row[ch] = (_Float16)0.0f;
}

// group_all for SA3: rows b*128+j = [l2_xyz (3) | l2_points (256)] pad to 288
__global__ void k_pack_groupall(const float* __restrict__ xyz, const float* __restrict__ pts,
                                int S, int c2, int Ck, _Float16* __restrict__ A, int b_) {
  long i = (long)blockIdx.x * blockDim.x + threadIdx.x;
  long tot = (long)b_ * S * Ck;
  if (i >= tot) return;
  int ch = (int)(i % Ck);
  long r = i / Ck;
  float v = 0.0f;
  if (ch < 3)            v = xyz[r * 3 + ch];
  else if (ch < 3 + c2)  v = pts[r * c2 + (ch - 3)];
  A[i] = (_Float16)v;
}

// generic 2-way concat: row r -> [X1[r] (c1) | X2[r/rep2] (c2)] pad to Ck
__global__ void k_pack_concat2(const float* __restrict__ X1, int c1,
                               const float* __restrict__ X2, int c2, int rep2,
                               int Ck, _Float16* __restrict__ A, long M) {
  long i = (long)blockIdx.x * blockDim.x + threadIdx.x;
  long tot = M * Ck;
  if (i >= tot) return;
  int ch = (int)(i % Ck);
  long r = i / Ck;
  float v = 0.0f;
  if (ch < c1)           v = X1[r * c1 + ch];
  else if (ch < c1 + c2) v = X2[(r / rep2) * (long)c2 + (ch - c1)];
  A[i] = (_Float16)v;
}

// FP1 input: [cls_oh (16) | l0_xyz (3) | l0_xyz again (3) | interp (128)] pad 160
__global__ void k_pack_fp1(const float* __restrict__ cls, const float* __restrict__ xyz,
                           const float* __restrict__ interp, int Ck,
                           _Float16* __restrict__ A, int n_, int b_) {
  long i = (long)blockIdx.x * blockDim.x + threadIdx.x;
  long tot = (long)b_ * n_ * Ck;
  if (i >= tot) return;
  int ch = (int)(i % Ck);
  long r = i / Ck;
  int b = (int)(r / n_);
  float v = 0.0f;
  if (ch < 16)       v = cls[(size_t)b * 16 + ch];
  else if (ch < 19)  v = xyz[r * 3 + (ch - 16)];
  else if (ch < 22)  v = xyz[r * 3 + (ch - 19)];
  else if (ch < 150) v = interp[r * 128 + (ch - 22)];
  A[i] = (_Float16)v;
}

// ---------------------------------------------------------------- weights
__global__ void k_pack_w(const float* __restrict__ w, _Float16* __restrict__ wp,
                         int cout, int cin, int coutPad, int Ck) {
  long i = (long)blockIdx.x * blockDim.x + threadIdx.x;
  long tot = (long)coutPad * Ck;
  if (i >= tot) return;
  int c = (int)(i % Ck);
  int o = (int)(i / Ck);
  wp[i] = (o < cout && c < cin) ? (_Float16)w[(size_t)o * cin + c] : (_Float16)0.0f;
}

__global__ void k_zero(float* __restrict__ p, int n) {
  for (int i = threadIdx.x; i < n; i += blockDim.x) p[i] = 0.0f;
}

// ---------------------------------------------------------------- WMMA GEMM
// Y[m, o] = sum_c A[m, c] * W[o, c] + bias[o].  A: M x Ck (f16), W: coutPad x Ck.
// One wave computes one 16x16 output tile; 8 waves/block cover 128 rows and
// share one tileN, so the 16 x Ck f16 weight tile is DMA-staged into LDS once
// per block by the Tensor Data Mover (tensor_load_to_lds, TENSORcnt-gated).
// A/B fragments use contiguous k (16*h + s); C/D layout per CDNA5 ISA 7.12.2 (wave32).
__global__ __launch_bounds__(256) void k_gemm_wmma(const _Float16* __restrict__ A,
                                                   const _Float16* __restrict__ W,
                                                   const float* __restrict__ bias,
                                                   float* __restrict__ Y,
                                                   long M, int Ck, int cout, int coutPad) {
  __shared__ _Float16 lw[16 * kCkMax];   // 40KB of the 320KB WGP LDS pool

  const int lane = threadIdx.x & 31;
  const int wv   = threadIdx.x >> 5;
  const long tileM = (long)blockIdx.x * 8 + wv;
  const int  tileN = blockIdx.y;
  const int  h  = lane >> 4;        // half-wave selector
  const int  ln = lane & 15;
  const long row = tileM * 16 + ln; // A row for this lane
  const int  col = tileN * 16 + ln; // output channel for this lane

  // ---- TDM: stage W rows [tileN*16, tileN*16+16) x Ck halves into LDS ----
  if (threadIdx.x < 32) {           // wave 0 issues the DMA (EXEC ignored by TDM)
    unsigned long long ga =
        (unsigned long long)(const void*)(W + (size_t)tileN * 16 * Ck);
    // Generic shared-space address: low 32 bits are the LDS byte offset
    // (flat LDS aperture mapping, ISA 10.2).
    unsigned ldsOff = (unsigned)(unsigned long long)(const void*)&lw[0];
    // D# group 0: count=1 | lds_addr | global_addr[56:0] | type=2 (ISA 8.3)
    v4u g0;
    g0.x = 0x1u;
    g0.y = ldsOff;
    g0.z = (unsigned)(ga & 0xFFFFFFFFull);
    g0.w = (unsigned)((ga >> 32) & 0x1FFFFFFull) | (2u << 30);
    // D# group 1: 2-D tensor, data_size=2B, dim0=Ck, dim1=16, tile=16 x Ck,
    // dim0_stride=Ck (ISA 8.4).
    v8i g1;
    g1[0] = (1 << 16);                              // data_size=1 -> 2 bytes
    g1[1] = (Ck & 0xFFFF) << 16;                    // tensor_dim0[15:0]
    g1[2] = (16 << 16) | ((Ck >> 16) & 0xFFFF);     // tensor_dim1 | dim0 hi
    g1[3] = (Ck & 0xFFFF) << 16;                    // tile_dim0
    g1[4] = 16;                                     // tile_dim1 (tile_dim2=0)
    g1[5] = Ck;                                     // tensor_dim0_stride lo
    g1[6] = 0;
    g1[7] = 0;
    v4i gz = {0, 0, 0, 0};                          // groups 2/3 unused (2-D)
    v8i gz8 = {0, 0, 0, 0, 0, 0, 0, 0};             // extra group (6-arg form)
    __builtin_amdgcn_tensor_load_to_lds(g0, g1, gz, gz, gz8, 0);
    __builtin_amdgcn_s_wait_tensorcnt(0);
  }
  __syncthreads();

  const _Float16* ap = A + (size_t)row * Ck;
  const _Float16* bp = lw + (size_t)ln * Ck + 16 * h;

  v8f acc = {};
  for (int kk = 0; kk < Ck; kk += 32) {
    __builtin_prefetch(ap + kk + 64, 0, 1);   // -> global_prefetch_b8
    v16h a, b;
    // A 16x32 f16 fragment: lanes0-15 hold K=0..15, lanes16-31 K=16..31,
    // contiguous (matches B order).
#pragma unroll
    for (int j = 0; j < 8; ++j) {
      const int K0 = 16 * h + 2 * j;
      a[2 * j]     = ap[kk + K0];
      a[2 * j + 1] = ap[kk + K0 + 1];
    }
    // B 32x16 f16 fragment from LDS: lanes0-15 hold K=0..15, lanes16-31
    // K=16..31, contiguous 16 halves per lane (-> ds_load_b128 pairs).
#pragma unroll
    for (int e = 0; e < 16; ++e) b[e] = bp[kk + e];
    acc = __builtin_amdgcn_wmma_f32_16x16x32_f16(false, a, false, b,
                                                 (short)0, acc, false, false);
  }
  if (col < cout) {
    const float bv = bias[col];
#pragma unroll
    for (int r = 0; r < 8; ++r) {           // C/D: VGPR r -> M = r + 8*h
      const long m = tileM * 16 + r + 8 * h;
      Y[(size_t)m * cout + col] = acc[r] + bv;
    }
  }
}

// ---------------------------------------------------------------- BN stats
__global__ __launch_bounds__(256) void k_bn_stats(const float* __restrict__ Y,
                                                  float* __restrict__ sums,
                                                  float* __restrict__ sqs,
                                                  long M, int cout) {
  __shared__ float ls[256], lq[256];
  int c = blockIdx.x;
  float s = 0.0f, q = 0.0f;
  for (long m = (long)blockIdx.y * blockDim.x + threadIdx.x; m < M;
       m += (long)gridDim.y * blockDim.x) {
    float v = Y[(size_t)m * cout + c];
    s += v; q += v * v;
  }
  ls[threadIdx.x] = s; lq[threadIdx.x] = q;
  __syncthreads();
  for (int st = 128; st > 0; st >>= 1) {
    if (threadIdx.x < st) { ls[threadIdx.x] += ls[threadIdx.x + st];
                            lq[threadIdx.x] += lq[threadIdx.x + st]; }
    __syncthreads();
  }
  if (threadIdx.x == 0) { atomicAdd(&sums[c], ls[0]); atomicAdd(&sqs[c], lq[0]); }
}

// ---------------------------------------------------------------- BN apply
__global__ void k_bn_apply(const float* __restrict__ Y,
                           const float* __restrict__ sums, const float* __restrict__ sqs,
                           const float* __restrict__ g, const float* __restrict__ beta,
                           float* __restrict__ out32, _Float16* __restrict__ out16,
                           long M, int cout) {
  long i = (long)blockIdx.x * blockDim.x + threadIdx.x;
  long tot = M * cout;
  if (i >= tot) return;
  int c = (int)(i % cout);
  float inv = 1.0f / (float)M;
  float mu = sums[c] * inv;
  float var = sqs[c] * inv - mu * mu;
  float v = (Y[i] - mu) * rsqrtf(var + BN_EPSF) * g[c] + beta[c];
  v = fmaxf(v, 0.0f);
  out32[i] = v;
  if (out16) out16[i] = (_Float16)v;
}

// ---------------------------------------------------------------- max pool
__global__ void k_maxpool(const float* __restrict__ Y, float* __restrict__ out,
                          long G, int K, int C) {
  long i = (long)blockIdx.x * blockDim.x + threadIdx.x;
  long tot = G * C;
  if (i >= tot) return;
  int c = (int)(i % C);
  long gq = i / C;
  float m = -3e38f;
  const float* p = Y + (size_t)gq * K * C + c;
  for (int k = 0; k < K; ++k) m = fmaxf(m, p[(size_t)k * C]);
  out[i] = m;
}

// ---------------------------------------------------------------- 3-NN
__global__ void k_3nn(const float* __restrict__ xyz1, const float* __restrict__ xyz2,
                      int n1, int n2, int* __restrict__ nidx, float* __restrict__ nw,
                      int b_) {
  long tid = (long)blockIdx.x * blockDim.x + threadIdx.x;
  long tot = (long)b_ * n1;
  if (tid >= tot) return;
  int b = (int)(tid / n1);
  const float* q = xyz1 + tid * 3;
  const float* p = xyz2 + (size_t)b * n2 * 3;
  float cx = q[0], cy = q[1], cz = q[2];
  float d0 = 3e38f, d1 = 3e38f, d2 = 3e38f;
  int i0 = 0, i1 = 0, i2 = 0;
  for (int j = 0; j < n2; ++j) {
    float dx = p[j * 3 + 0] - cx, dy = p[j * 3 + 1] - cy, dz = p[j * 3 + 2] - cz;
    float d = dx * dx + dy * dy + dz * dz;
    if (d < d0)      { d2 = d1; i2 = i1; d1 = d0; i1 = i0; d0 = d; i0 = j; }
    else if (d < d1) { d2 = d1; i2 = i1; d1 = d;  i1 = j; }
    else if (d < d2) { d2 = d;  i2 = j; }
  }
  float w0 = 1.0f / (d0 + 1e-8f), w1 = 1.0f / (d1 + 1e-8f), w2 = 1.0f / (d2 + 1e-8f);
  float ws = w0 + w1 + w2;
  nidx[tid * 3 + 0] = i0; nidx[tid * 3 + 1] = i1; nidx[tid * 3 + 2] = i2;
  nw[tid * 3 + 0] = w0 / ws; nw[tid * 3 + 1] = w1 / ws; nw[tid * 3 + 2] = w2 / ws;
}

__global__ void k_interp(const float* __restrict__ p2, const int* __restrict__ nidx,
                         const float* __restrict__ nw, float* __restrict__ out,
                         int n1, int n2, int C, int b_) {
  long i = (long)blockIdx.x * blockDim.x + threadIdx.x;
  long tot = (long)b_ * n1 * C;
  if (i >= tot) return;
  int c = (int)(i % C);
  long r = i / C;
  int b = (int)(r / n1);
  const int* id = nidx + r * 3;
  const float* w = nw + r * 3;
  const float* base = p2 + (size_t)b * n2 * C;
  out[i] = w[0] * base[(size_t)id[0] * C + c] +
           w[1] * base[(size_t)id[1] * C + c] +
           w[2] * base[(size_t)id[2] * C + c];
}

// ---------------------------------------------------------------- softmax
__global__ void k_logsoftmax(const float* __restrict__ Y, float* __restrict__ out,
                             long rows, int C) {
  long r = (long)blockIdx.x * blockDim.x + threadIdx.x;
  if (r >= rows) return;
  const float* p = Y + (size_t)r * C;
  float m = -3e38f;
  for (int c = 0; c < C; ++c) m = fmaxf(m, p[c]);
  float s = 0.0f;
  for (int c = 0; c < C; ++c) s += expf(p[c] - m);
  float ls = logf(s);
  float* o = out + (size_t)r * C;
  for (int c = 0; c < C; ++c) o[c] = p[c] - m - ls;
}

__global__ void k_copy(const float* __restrict__ src, float* __restrict__ dst, long n) {
  long i = (long)blockIdx.x * blockDim.x + threadIdx.x;
  if (i < n) dst[i] = src[i];
}

// ===========================================================================
// Host orchestration
// ===========================================================================
struct LP { const float *w, *b, *g, *beta; int cin, cout; };

extern "C" void kernel_launch(void* const* d_in, const int* in_sizes, int n_in,
                              void* d_out, int out_size, void* d_ws, size_t ws_size,
                              hipStream_t stream) {
  (void)in_sizes; (void)n_in; (void)out_size; (void)ws_size;
  // Input flattening (insertion order of setup_inputs pytree):
  // 0:xyz 1:cls_label then sa1(3x{w,b,g,beta}) sa2 sa3 fp3(2) fp2(2) fp1(3)
  // conv1{w,b} bn1{g,beta} conv2{w,b}
  const float* in_xyz = (const float*)d_in[0];
  const float* cls    = (const float*)d_in[1];
  int ii = 2;
  auto next4 = [&](int cin, int cout) {
    LP l{ (const float*)d_in[ii], (const float*)d_in[ii + 1],
          (const float*)d_in[ii + 2], (const float*)d_in[ii + 3], cin, cout };
    ii += 4; return l;
  };
  LP sa1[3] = { next4(6, 64),    next4(64, 64),   next4(64, 128) };
  LP sa2[3] = { next4(131, 128), next4(128, 128), next4(128, 256) };
  LP sa3[3] = { next4(259, 256), next4(256, 512), next4(512, 1024) };
  LP fp3[2] = { next4(1280, 256), next4(256, 256) };
  LP fp2[2] = { next4(384, 256),  next4(256, 128) };
  LP fp1[3] = { next4(150, 128),  next4(128, 128), next4(128, 128) };
  const float* c1w = (const float*)d_in[ii++]; const float* c1b = (const float*)d_in[ii++];
  const float* bn1g = (const float*)d_in[ii++]; const float* bn1b = (const float*)d_in[ii++];
  const float* c2w = (const float*)d_in[ii++]; const float* c2b = (const float*)d_in[ii++];

  // ---- workspace layout (bump allocator; requires ~504 MB scratch) ----
  char* ws = (char*)d_ws; size_t off = 0;
  auto alloc = [&](size_t bytes) { void* p = ws + off; off += (bytes + 255) & ~(size_t)255; return p; };
  float* l0_xyz = (float*)alloc((size_t)kB * kN * 3 * 4);
  float* l1_xyz = (float*)alloc((size_t)kB * 512 * 3 * 4);
  float* l2_xyz = (float*)alloc((size_t)kB * 128 * 3 * 4);
  float* l1_pts = (float*)alloc((size_t)kB * 512 * 128 * 4);
  float* l2_pts = (float*)alloc((size_t)kB * 128 * 256 * 4);
  float* l3_pts = (float*)alloc((size_t)kB * 1024 * 4);
  float* l2p_fp = (float*)alloc((size_t)kB * 128 * 256 * 4);
  float* l1p_fp = (float*)alloc((size_t)kB * 512 * 128 * 4);
  float* interp = (float*)alloc((size_t)kB * kN * 128 * 4);
  int*   fidx   = (int*)alloc((size_t)kB * 512 * 4);
  int*   qidx   = (int*)alloc((size_t)kB * 16384 * 4);
  int*   nidx   = (int*)alloc((size_t)kB * kN * 3 * 4);
  float* nww    = (float*)alloc((size_t)kB * kN * 3 * 4);
  float* stats  = (float*)alloc(2048 * 4);
  _Float16* Wp  = (_Float16*)alloc((size_t)2097152 * 2);           // max 1024x512 (SA3 L3)
  size_t aHalves = (size_t)262144 * 160;                           // SA2 L1 input
  _Float16* A0  = (_Float16*)alloc(aHalves * 2);
  _Float16* A1  = (_Float16*)alloc(aHalves * 2);
  float* Yb     = (float*)alloc((size_t)524288 * 128 * 4);         // SA1 L3 output

  auto gb = [](long n) { return (unsigned)((n + 255) / 256); };

  // GEMM + bias + global BN + ReLU -> (f32 out32, f16 Aout)
  auto run_layer = [&](const _Float16* Ain, _Float16* Aout, const LP& l,
                       const float* g, const float* beta, long M, int Ck, float* out32) {
    int coutPad = (l.cout + 15) & ~15;
    long wtot = (long)coutPad * Ck;
    k_pack_w<<<gb(wtot), 256, 0, stream>>>(l.w, Wp, l.cout, l.cin, coutPad, Ck);
    k_gemm_wmma<<<dim3((unsigned)(M / 128), coutPad / 16), 256, 0, stream>>>(
        Ain, Wp, l.b, Yb, M, Ck, l.cout, coutPad);
    k_zero<<<1, 256, 0, stream>>>(stats, 2048);
    k_bn_stats<<<dim3(l.cout, 128), 256, 0, stream>>>(Yb, stats, stats + 1024, M, l.cout);
    k_bn_apply<<<gb(M * l.cout), 256, 0, stream>>>(Yb, stats, stats + 1024, g, beta,
                                                   out32, Aout, M, l.cout);
  };

  // ---- l0 ----
  k_transpose<<<gb((long)kB * kN * 3), 256, 0, stream>>>(in_xyz, l0_xyz, kB, kN);

  // ---- SA1: 2048 -> 512 pts, K=32, cin 6 (pad 32), widths 64,64,128 ----
  k_fps<<<kB, 256, 0, stream>>>(l0_xyz, kN, 512, fidx);
  k_gather_xyz<<<gb((long)kB * 512 * 3), 256, 0, stream>>>(l0_xyz, fidx, l1_xyz, kN, 512, kB);
  k_query_ball<<<(unsigned)((long)kB * 512 / 8), 256, 0, stream>>>(
      l0_xyz, l1_xyz, kN, 512, 0.2f * 0.2f, 32, qidx);
  {
    long M = (long)kB * 512 * 32;
    k_group<<<gb(M), 256, 0, stream>>>(l0_xyz, l0_xyz, 3, qidx, l1_xyz, kN, 512, 32, 32, A0, kB);
    run_layer(A0, A1, sa1[0], sa1[0].g, sa1[0].beta, M, 32, Yb);
    run_layer(A1, A0, sa1[1], sa1[1].g, sa1[1].beta, M, 64, Yb);
    run_layer(A0, A1, sa1[2], sa1[2].g, sa1[2].beta, M, 64, Yb);
    k_maxpool<<<gb((long)kB * 512 * 128), 256, 0, stream>>>(Yb, l1_pts, (long)kB * 512, 32, 128);
  }

  // ---- SA2: 512 -> 128 pts, K=64, cin 131 (pad 160), widths 128,128,256 ----
  k_fps<<<kB, 256, 0, stream>>>(l1_xyz, 512, 128, fidx);
  k_gather_xyz<<<gb((long)kB * 128 * 3), 256, 0, stream>>>(l1_xyz, fidx, l2_xyz, 512, 128, kB);
  k_query_ball<<<(unsigned)((long)kB * 128 / 8), 256, 0, stream>>>(
      l1_xyz, l2_xyz, 512, 128, 0.4f * 0.4f, 64, qidx);
  {
    long M = (long)kB * 128 * 64;
    k_group<<<gb(M), 256, 0, stream>>>(l1_xyz, l1_pts, 128, qidx, l2_xyz, 512, 128, 64, 160, A0, kB);
    run_layer(A0, A1, sa2[0], sa2[0].g, sa2[0].beta, M, 160, Yb);
    run_layer(A1, A0, sa2[1], sa2[1].g, sa2[1].beta, M, 128, Yb);
    run_layer(A0, A1, sa2[2], sa2[2].g, sa2[2].beta, M, 128, Yb);
    k_maxpool<<<gb((long)kB * 128 * 256), 256, 0, stream>>>(Yb, l2_pts, (long)kB * 128, 64, 256);
  }

  // ---- SA3: group_all, rows = B*128, cin 259 (pad 288), widths 256,512,1024 ----
  {
    long M = (long)kB * 128;
    k_pack_groupall<<<gb(M * 288), 256, 0, stream>>>(l2_xyz, l2_pts, 128, 256, 288, A0, kB);
    run_layer(A0, A1, sa3[0], sa3[0].g, sa3[0].beta, M, 288, Yb);
    run_layer(A1, A0, sa3[1], sa3[1].g, sa3[1].beta, M, 256, Yb);
    run_layer(A0, A1, sa3[2], sa3[2].g, sa3[2].beta, M, 512, Yb);
    k_maxpool<<<gb((long)kB * 1024), 256, 0, stream>>>(Yb, l3_pts, (long)kB, 128, 1024);
  }

  // ---- FP3: broadcast l3 (s==1), concat 256+1024 = 1280 -> 256,256 ----
  {
    long M = (long)kB * 128;
    k_pack_concat2<<<gb(M * 1280), 256, 0, stream>>>(l2_pts, 256, l3_pts, 1024, 128, 1280, A0, M);
    run_layer(A0, A1, fp3[0], fp3[0].g, fp3[0].beta, M, 1280, Yb);
    run_layer(A1, A0, fp3[1], fp3[1].g, fp3[1].beta, M, 256, l2p_fp);
  }

  // ---- FP2: 3-NN interp (512 <- 128), concat 128+256 = 384 -> 256,128 ----
  {
    long M = (long)kB * 512;
    k_3nn<<<gb(M), 256, 0, stream>>>(l1_xyz, l2_xyz, 512, 128, nidx, nww, kB);
    k_interp<<<gb(M * 256), 256, 0, stream>>>(l2p_fp, nidx, nww, interp, 512, 128, 256, kB);
    k_pack_concat2<<<gb(M * 384), 256, 0, stream>>>(l1_pts, 128, interp, 256, 1, 384, A0, M);
    run_layer(A0, A1, fp2[0], fp2[0].g, fp2[0].beta, M, 384, Yb);
    run_layer(A1, A0, fp2[1], fp2[1].g, fp2[1].beta, M, 256, l1p_fp);
  }

  // ---- FP1: 3-NN interp (2048 <- 512), concat 16+3+3+128 = 150 (pad 160) ----
  long Mn = (long)kB * kN;
  {
    k_3nn<<<gb(Mn), 256, 0, stream>>>(l0_xyz, l1_xyz, kN, 512, nidx, nww, kB);
    k_interp<<<gb(Mn * 128), 256, 0, stream>>>(l1p_fp, nidx, nww, interp, kN, 512, 128, kB);
    k_pack_fp1<<<gb(Mn * 160), 256, 0, stream>>>(cls, l0_xyz, interp, 160, A0, kN, kB);
    run_layer(A0, A1, fp1[0], fp1[0].g, fp1[0].beta, Mn, 160, Yb);
    run_layer(A1, A0, fp1[1], fp1[1].g, fp1[1].beta, Mn, 128, Yb);
    run_layer(A0, A1, fp1[2], fp1[2].g, fp1[2].beta, Mn, 128, Yb);
  }

  // ---- head: conv1 (128->128, bn1, relu) -> conv2 (128->50) -> log_softmax ----
  {
    LP lc1{ c1w, c1b, bn1g, bn1b, 128, 128 };
    run_layer(A1, A0, lc1, bn1g, bn1b, Mn, 128, Yb);
    // conv2: GEMM + bias only (coutPad 64, stores 50 channels)
    int coutPad = 64, Ck = 128;
    k_pack_w<<<gb((long)coutPad * Ck), 256, 0, stream>>>(c2w, Wp, kCls, 128, coutPad, Ck);
    k_gemm_wmma<<<dim3((unsigned)(Mn / 128), coutPad / 16), 256, 0, stream>>>(
        A0, Wp, c2b, Yb, Mn, Ck, kCls, coutPad);
    k_logsoftmax<<<gb(Mn), 256, 0, stream>>>(Yb, (float*)d_out, Mn, kCls);
  }

  // ---- second output: l3_points transposed (B,1,1024)->(B,1024,1) == flat copy
  k_copy<<<gb((long)kB * 1024), 256, 0, stream>>>(
      l3_pts, (float*)d_out + (size_t)Mn * kCls, (long)kB * 1024);
}
